// ReinforceRL4CVRP_15315853377622
// MI455X (gfx1250) — compile-verified, matrix operand order checked by reading, not run on hardware
//
#include <hip/hip_runtime.h>
#include <hip/hip_bf16.h>
#include <math.h>

typedef __attribute__((ext_vector_type(16))) __bf16 v16bf;
typedef __attribute__((ext_vector_type(8)))  __bf16 v8bf;
typedef __attribute__((ext_vector_type(8)))  float  v8f;

constexpr int kB = 128;    // batch
constexpr int kN = 20;     // nodes
constexpr int kH = 128;    // hidden
constexpr int kS = 2;      // static features
constexpr int kSteps = 100;
constexpr int kMB = 16;    // batches per decode workgroup -> 8 WGs

// ---------------------------------------------------------------------------
// Pack weights into bf16 WMMA B-fragment layout (K=32 x N=16 per tile).
// Lane l -> column n = nt*16 + (l&15). Element e (of 16 per lane):
//   lanes 0-15 : e in [0,8) -> k = kt*32 +  0 + e ; e in [8,16) -> k = kt*32 + 16 + (e-8)
//   lanes 16-31: e in [0,8) -> k = kt*32 +  8 + e ; e in [8,16) -> k = kt*32 + 24 + (e-8)
// (mirror of the documented 16-bit A-matrix layout). B[k][n] = W[n][k].
// ---------------------------------------------------------------------------
__global__ void rl4cvrp_pack_kernel(const float* __restrict__ Wih,
                                    const float* __restrict__ Whh,
                                    const float* __restrict__ attnW,
                                    const float* __restrict__ ptrW,
                                    const float* __restrict__ dyn,
                                    float* __restrict__ out,
                                    __bf16* __restrict__ WihP,
                                    __bf16* __restrict__ WhhP,
                                    __bf16* __restrict__ Wa4P,
                                    __bf16* __restrict__ Wp2P) {
  const int tid = blockIdx.x * blockDim.x + threadIdx.x;
  const int nth = gridDim.x * blockDim.x;

  // pass-through output: dynamic (B,S,N)
  for (int i = tid; i < kB * kS * kN; i += nth) out[i] = dyn[i];

  // GRU weights: 24 n-tiles (3H) x 4 k-tiles (H)
  for (int i = tid; i < 24 * 4 * 32 * 16; i += nth) {
    const int e = i & 15, lane = (i >> 4) & 31, kt = (i >> 9) & 3, nt = i >> 11;
    const int n = nt * 16 + (lane & 15);
    const int base = (lane < 16) ? ((e < 8) ? 0 : 16) : ((e < 8) ? 8 : 24);
    const int k = kt * 32 + base + (e & 7);
    WihP[i] = (__bf16)Wih[n * kH + k];
    WhhP[i] = (__bf16)Whh[n * kH + k];
  }
  // Wa4 = attn_W[:, 3H:4H], Wp2 = ptr_W[:, H:2H]: 8 n-tiles x 4 k-tiles
  for (int i = tid; i < 8 * 4 * 32 * 16; i += nth) {
    const int e = i & 15, lane = (i >> 4) & 31, kt = (i >> 9) & 3, nt = i >> 11;
    const int n = nt * 16 + (lane & 15);
    const int base = (lane < 16) ? ((e < 8) ? 0 : 16) : ((e < 8) ? 8 : 24);
    const int k = kt * 32 + base + (e & 7);
    Wa4P[i] = (__bf16)attnW[n * (4 * kH) + 3 * kH + k];
    Wp2P[i] = (__bf16)ptrW[n * (2 * kH) + kH + k];
  }
}

// ---------------------------------------------------------------------------
// Step-invariant precompute, one workgroup per batch:
//   adj_h/stat_h/dyn_h (H,N) in LDS, then attn_pre / ptr_pre (B,H,N) to ws.
// ---------------------------------------------------------------------------
__global__ __launch_bounds__(256) void rl4cvrp_precompute_kernel(
    const float* __restrict__ adj, const float* __restrict__ stat,
    const float* __restrict__ dynv, const float* __restrict__ depot,
    const float* __restrict__ W_adj, const float* __restrict__ b_adj,
    const float* __restrict__ W_stat, const float* __restrict__ b_stat,
    const float* __restrict__ W_dyn, const float* __restrict__ b_dyn,
    const float* __restrict__ attnW, const float* __restrict__ ptrW,
    float* __restrict__ statH, float* __restrict__ attnPre,
    float* __restrict__ ptrPre, float* __restrict__ logMask) {
  __shared__ float adj_s[kH][kN];
  __shared__ float st_s[kH][kN];
  __shared__ float dy_s[kH][kN];

  const int b = blockIdx.x;
  const int tid = threadIdx.x;

  for (int i = tid; i < kH * kN; i += blockDim.x) {
    const int h = i / kN, n = i % kN;
    float sa = b_adj[h];
    for (int c = 0; c < kN; ++c)
      sa += W_adj[h * (kN + 1) + c] * adj[b * kN * kN + c * kN + n];
    sa += W_adj[h * (kN + 1) + kN] * depot[b * kN + n];
    adj_s[h][n] = sa;

    const float ss = b_stat[h] + W_stat[h * kS + 0] * stat[b * kS * kN + n] +
                     W_stat[h * kS + 1] * stat[b * kS * kN + kN + n];
    st_s[h][n] = ss;
    statH[b * kH * kN + h * kN + n] = ss;

    dy_s[h][n] = b_dyn[h] + W_dyn[h * kS + 0] * dynv[b * kS * kN + n] +
                 W_dyn[h * kS + 1] * dynv[b * kS * kN + kN + n];
  }
  for (int i = tid; i < kN; i += blockDim.x)
    logMask[b * kN + i] = logf(depot[b * kN + i]);
  __syncthreads();

  for (int i = tid; i < kH * kN; i += blockDim.x) {
    const int h = i / kN, n = i % kN;
    float ap = 0.f, pp = 0.f;
    const float* aw = attnW + (size_t)h * 4 * kH;
    const float* pw = ptrW + (size_t)h * 2 * kH;
    for (int k = 0; k < kH; ++k) {
      ap += aw[k] * adj_s[k][n] + aw[kH + k] * st_s[k][n] +
            aw[2 * kH + k] * dy_s[k][n];
      pp += pw[k] * st_s[k][n];
    }
    attnPre[b * kH * kN + h * kN + n] = ap;
    ptrPre[b * kH * kN + h * kN + n] = pp;
  }
}

// ---------------------------------------------------------------------------
// Decode: 8 persistent workgroups x 256 threads (8 wave32), 100 steps.
// ---------------------------------------------------------------------------
__device__ __forceinline__ v16bf load_a_frag(const __bf16* __restrict__ row,
                                             int lane, int kt) {
  // 16-bit A fragment (16x32): lanes 0-15 take k {0..7,16..23}, lanes 16-31
  // take k {8..15,24..31} of this K-tile.
  const int lo = (lane < 16) ? 0 : 8;
  union { v16bf v; v8bf h[2]; } u;
  u.h[0] = *(const v8bf*)(row + kt * 32 + lo);
  u.h[1] = *(const v8bf*)(row + kt * 32 + lo + 16);
  return u.v;
}

__device__ __forceinline__ v16bf load_b_frag(const __bf16* __restrict__ Wp,
                                             int nt, int kt, int lane) {
  return *(const v16bf*)(Wp + ((((size_t)nt * 4 + kt) * 32 + lane) << 4));
}

__device__ __forceinline__ float sigm(float x) { return 1.f / (1.f + __expf(-x)); }

__global__ __launch_bounds__(256) void rl4cvrp_decode_kernel(
    const float* __restrict__ attnPre, const float* __restrict__ ptrPre,
    const float* __restrict__ statH, const float* __restrict__ logMask,
    const __bf16* __restrict__ WihP, const __bf16* __restrict__ WhhP,
    const __bf16* __restrict__ Wa4P, const __bf16* __restrict__ Wp2P,
    const float* __restrict__ W_dec, const float* __restrict__ b_dec,
    const float* __restrict__ bih, const float* __restrict__ bhh,
    const float* __restrict__ attn_v, const float* __restrict__ ptr_v,
    const float* __restrict__ stat_g, const float* __restrict__ x0,
    float* __restrict__ out) {
  __shared__ __bf16 dec_bf[kMB][kH];
  __shared__ __bf16 h_bf[kMB][kH];
  __shared__ float  h_f[kMB][kH];
  __shared__ float  q_f[kMB][kH];
  __shared__ __bf16 ctx_bf[kMB][kH];
  __shared__ float  a_l[kMB][kN];
  __shared__ float  p_l[kMB][kN];
  __shared__ float  attnw[kMB][kN];
  __shared__ float  dec_in[kMB][kS];

  const int tid = threadIdx.x;
  const int lane = tid & 31;
  const int w = tid >> 5;           // wave id 0..7 -> owns columns [16w,16w+16)
  const int l15 = lane & 15;
  const int mup = (lane >> 4) * 8;  // C-layout: VGPR r -> row m = mup + r
  const int wgb = blockIdx.x * kMB; // first global batch of this WG

  for (int i = tid; i < kMB * kH; i += 256) {
    (&h_f[0][0])[i] = 0.f;
    (&h_bf[0][0])[i] = (__bf16)0.f;
  }
  if (tid < kMB * kS) dec_in[tid >> 1][tid & 1] = x0[tid & 1];
  __syncthreads();

  // per-lane GRU biases for this wave's column slice (r/z/n gate slices)
  const float bir = bih[w * 16 + l15],            bhr = bhh[w * 16 + l15];
  const float biz = bih[kH + w * 16 + l15],       bhz = bhh[kH + w * 16 + l15];
  const float bin_ = bih[2 * kH + w * 16 + l15],  bhn = bhh[2 * kH + w * 16 + l15];

  for (int step = 0; step < kSteps; ++step) {
    // ---- stage 0: dec_h = dec_in @ W_dec^T + b_dec -> bf16 A source
    for (int i = tid; i < kMB * kH; i += 256) {
      const int m = i >> 7, j = i & 127;
      const float v = b_dec[j] + dec_in[m][0] * W_dec[j * kS] +
                      dec_in[m][1] * W_dec[j * kS + 1];
      dec_bf[m][j] = (__bf16)v;
    }
    __syncthreads();

    // ---- stage 1: GRU GEMMs, wave w owns gate columns 16w..16w+15
    v8f cir = {}, ciz = {}, cin = {}, chr_ = {}, chz = {}, chn = {};
    {
      const __bf16* arow = &dec_bf[l15][0];
      const __bf16* hrow = &h_bf[l15][0];
#pragma unroll
      for (int kt = 0; kt < 4; ++kt) {
        const v16bf ad = load_a_frag(arow, lane, kt);
        const v16bf ah = load_a_frag(hrow, lane, kt);
        const v16bf b0 = load_b_frag(WihP, w, kt, lane);
        const v16bf b1 = load_b_frag(WihP, 8 + w, kt, lane);
        const v16bf b2 = load_b_frag(WihP, 16 + w, kt, lane);
        const v16bf b3 = load_b_frag(WhhP, w, kt, lane);
        const v16bf b4 = load_b_frag(WhhP, 8 + w, kt, lane);
        const v16bf b5 = load_b_frag(WhhP, 16 + w, kt, lane);
        cir = __builtin_amdgcn_wmma_f32_16x16x32_bf16(false, ad, false, b0, (short)0, cir, false, false);
        ciz = __builtin_amdgcn_wmma_f32_16x16x32_bf16(false, ad, false, b1, (short)0, ciz, false, false);
        cin = __builtin_amdgcn_wmma_f32_16x16x32_bf16(false, ad, false, b2, (short)0, cin, false, false);
        chr_ = __builtin_amdgcn_wmma_f32_16x16x32_bf16(false, ah, false, b3, (short)0, chr_, false, false);
        chz = __builtin_amdgcn_wmma_f32_16x16x32_bf16(false, ah, false, b4, (short)0, chz, false, false);
        chn = __builtin_amdgcn_wmma_f32_16x16x32_bf16(false, ah, false, b5, (short)0, chn, false, false);
      }
    }
    __syncthreads();  // all waves done reading h_bf/dec_bf before h update

    {
      const int col = w * 16 + l15;
#pragma unroll
      for (int r = 0; r < 8; ++r) {
        const int m = mup + r;
        const float rg = sigm(cir[r] + bir + chr_[r] + bhr);
        const float zg = sigm(ciz[r] + biz + chz[r] + bhz);
        const float ng = tanhf(cin[r] + bin_ + rg * (chn[r] + bhn));
        const float hnew = (1.f - zg) * ng + zg * h_f[m][col];
        h_f[m][col] = hnew;
        h_bf[m][col] = (__bf16)hnew;
      }
    }
    __syncthreads();

    // ---- stage 2: q = h_new @ Wa4^T (wave w -> n-tile w)
    {
      v8f cq = {};
#pragma unroll
      for (int kt = 0; kt < 4; ++kt) {
        const v16bf ah = load_a_frag(&h_bf[l15][0], lane, kt);
        const v16bf bb = load_b_frag(Wa4P, w, kt, lane);
        cq = __builtin_amdgcn_wmma_f32_16x16x32_bf16(false, ah, false, bb, (short)0, cq, false, false);
      }
      const int col = w * 16 + l15;
#pragma unroll
      for (int r = 0; r < 8; ++r) q_f[mup + r][col] = cq[r];
    }
    __syncthreads();

    // ---- stage 3: a[b,n] = sum_h attn_v[h]*tanh(attn_pre + q) (wave dot)
    for (int t = 0; t < (kMB * kN) / 8; ++t) {
      const int p = w * ((kMB * kN) / 8) + t;
      const int bb = p / kN, n = p % kN;
      float s = 0.f;
      const float* ap = attnPre + ((size_t)(wgb + bb) * kH) * kN + n;
      for (int kk = lane; kk < kH; kk += 32)
        s += attn_v[kk] * tanhf(ap[(size_t)kk * kN] + q_f[bb][kk]);
      for (int o = 16; o > 0; o >>= 1) s += __shfl_xor(s, o, 32);
      if (lane == 0) a_l[bb][n] = s;
    }
    __syncthreads();

    // ---- stage 4: softmax over N
    if (tid < kMB) {
      float mx = -1e30f;
      for (int n = 0; n < kN; ++n) mx = fmaxf(mx, a_l[tid][n]);
      float sum = 0.f;
      for (int n = 0; n < kN; ++n) {
        const float e = __expf(a_l[tid][n] - mx);
        attnw[tid][n] = e;
        sum += e;
      }
      const float inv = 1.f / sum;
      for (int n = 0; n < kN; ++n) attnw[tid][n] *= inv;
    }
    __syncthreads();

    // ---- stage 5: ctx[b,h] = sum_n attn[b,n] * stat_h[b,h,n]
    for (int i = tid; i < kMB * kH; i += 256) {
      const int m = i >> 7, hh = i & 127;
      const float* sh = statH + ((size_t)(wgb + m) * kH + hh) * kN;
      float s = 0.f;
      for (int n = 0; n < kN; ++n) s += attnw[m][n] * sh[n];
      ctx_bf[m][hh] = (__bf16)s;
    }
    __syncthreads();

    // ---- stage 6: u = ctx @ Wp2^T (reuse q_f)
    {
      v8f cu = {};
#pragma unroll
      for (int kt = 0; kt < 4; ++kt) {
        const v16bf ac = load_a_frag(&ctx_bf[l15][0], lane, kt);
        const v16bf bb = load_b_frag(Wp2P, w, kt, lane);
        cu = __builtin_amdgcn_wmma_f32_16x16x32_bf16(false, ac, false, bb, (short)0, cu, false, false);
      }
      const int col = w * 16 + l15;
#pragma unroll
      for (int r = 0; r < 8; ++r) q_f[mup + r][col] = cu[r];
    }
    __syncthreads();

    // ---- stage 7: p[b,n] = sum_h ptr_v[h]*tanh(ptr_pre + u)
    for (int t = 0; t < (kMB * kN) / 8; ++t) {
      const int p = w * ((kMB * kN) / 8) + t;
      const int bb = p / kN, n = p % kN;
      float s = 0.f;
      const float* pp = ptrPre + ((size_t)(wgb + bb) * kH) * kN + n;
      for (int kk = lane; kk < kH; kk += 32)
        s += ptr_v[kk] * tanhf(pp[(size_t)kk * kN] + q_f[bb][kk]);
      for (int o = 16; o > 0; o >>= 1) s += __shfl_xor(s, o, 32);
      if (lane == 0) p_l[bb][n] = s;
    }
    __syncthreads();

    // ---- stage 8: masked log-softmax argmax, outputs, next dec_in
    if (tid < kMB) {
      const int bg = wgb + tid;
      float best = -1e30f;
      int bi = 0;
      float vals[kN];
      for (int n = 0; n < kN; ++n) {
        const float v = p_l[tid][n] + logMask[bg * kN + n];
        vals[n] = v;
        if (v > best) { best = v; bi = n; }
      }
      float sum = 0.f;
      for (int n = 0; n < kN; ++n) sum += __expf(vals[n] - best);
      const float logp = -__logf(sum);  // max(lsm) = best - (best + log(sum))
      out[kB * kS * kN + (size_t)bg * kSteps + step] = (float)bi;
      out[kB * kS * kN + kB * kSteps + (size_t)bg * kSteps + step] = logp;
      dec_in[tid][0] = stat_g[(size_t)bg * kS * kN + bi];
      dec_in[tid][1] = stat_g[(size_t)bg * kS * kN + kN + bi];
    }
    __syncthreads();
  }
}

// ---------------------------------------------------------------------------
extern "C" void kernel_launch(void* const* d_in, const int* in_sizes, int n_in,
                              void* d_out, int out_size, void* d_ws, size_t ws_size,
                              hipStream_t stream) {
  const float* adj    = (const float*)d_in[0];
  const float* stat   = (const float*)d_in[1];
  const float* dynv   = (const float*)d_in[2];
  const float* depot  = (const float*)d_in[3];
  const float* W_adj  = (const float*)d_in[4];
  const float* b_adj  = (const float*)d_in[5];
  const float* W_stat = (const float*)d_in[6];
  const float* b_stat = (const float*)d_in[7];
  const float* W_dyn  = (const float*)d_in[8];
  const float* b_dyn  = (const float*)d_in[9];
  const float* W_dec  = (const float*)d_in[10];
  const float* b_dec  = (const float*)d_in[11];
  const float* Wih    = (const float*)d_in[12];
  const float* Whh    = (const float*)d_in[13];
  const float* bih    = (const float*)d_in[14];
  const float* bhh    = (const float*)d_in[15];
  const float* attn_v = (const float*)d_in[16];
  const float* attnW  = (const float*)d_in[17];
  const float* ptr_v  = (const float*)d_in[18];
  const float* ptrW   = (const float*)d_in[19];
  const float* x0     = (const float*)d_in[20];
  float* out = (float*)d_out;

  // workspace carve-up (256B aligned chunks)
  char* ws = (char*)d_ws;
  size_t off = 0;
  auto carve = [&](size_t bytes) -> char* {
    char* p = ws + off;
    off += (bytes + 255) & ~(size_t)255;
    return p;
  };
  float*  statH   = (float*)carve((size_t)kB * kH * kN * sizeof(float));
  float*  attnPre = (float*)carve((size_t)kB * kH * kN * sizeof(float));
  float*  ptrPre  = (float*)carve((size_t)kB * kH * kN * sizeof(float));
  float*  logMask = (float*)carve((size_t)kB * kN * sizeof(float));
  __bf16* WihP    = (__bf16*)carve((size_t)24 * 4 * 32 * 16 * sizeof(__bf16));
  __bf16* WhhP    = (__bf16*)carve((size_t)24 * 4 * 32 * 16 * sizeof(__bf16));
  __bf16* Wa4P    = (__bf16*)carve((size_t)8 * 4 * 32 * 16 * sizeof(__bf16));
  __bf16* Wp2P    = (__bf16*)carve((size_t)8 * 4 * 32 * 16 * sizeof(__bf16));

  rl4cvrp_pack_kernel<<<96, 256, 0, stream>>>(Wih, Whh, attnW, ptrW, dynv, out,
                                              WihP, WhhP, Wa4P, Wp2P);
  rl4cvrp_precompute_kernel<<<kB, 256, 0, stream>>>(
      adj, stat, dynv, depot, W_adj, b_adj, W_stat, b_stat, W_dyn, b_dyn,
      attnW, ptrW, statH, attnPre, ptrPre, logMask);
  rl4cvrp_decode_kernel<<<kB / kMB, 256, 0, stream>>>(
      attnPre, ptrPre, statH, logMask, WihP, WhhP, Wa4P, Wp2P, W_dec, b_dec,
      bih, bhh, attn_v, ptr_v, stat, x0, out);
}
